// LSTMRNN_5025111737012
// MI455X (gfx1250) — matrix operand order repaired and checked by reading, not compiled, not run on hardware
//
#include <hip/hip_runtime.h>
#include <hip/hip_bf16.h>

// LSTM: V=50000, L=3, H=64, E=64, B=256, T=512, O=3
#define LL 3
#define HH 64
#define EE 64
#define BB 256
#define TT 512
#define OO 3
#define G4H 256   // 4*H

typedef __attribute__((ext_vector_type(16))) _Float16 v16h;
typedef __attribute__((ext_vector_type(8)))  _Float16 v8h;
typedef __attribute__((ext_vector_type(4)))  _Float16 v4h;
typedef __attribute__((ext_vector_type(8)))  float    v8f;
typedef __attribute__((ext_vector_type(4)))  float    v4f;

// ---------------------------------------------------------------------------
// WMMA operand staging (wave32; A/B in this code's contiguous K order, C/D per CDNA5 ISA)
//
// A 16x32 f16 (MxK): lane -> M = lane&15, K = (lane&16) + j
// B 32x16 f16 (KxN): lane -> N = lane&15, K = (lane&16) + j
// C/D 16x16 f32:     lane -> N = lane&15, M = (lane&16?8:0) + j
// All row strides here are 64 elements (E=H=64).
// ---------------------------------------------------------------------------

__device__ __forceinline__ v16h loadA16(const _Float16* s, int m0, int k0, int lane) {
    int row = m0 + (lane & 15);
    int kb  = k0 + ((lane & 16) ? 16 : 0);
    const _Float16* p = s + row * 64 + kb;
    return *(const v16h*)p;           // K = kb .. kb+15
}

// B = W^T where W is row-major [N rows][64 cols] f16 (LDS or global)
__device__ __forceinline__ v16h loadBT16(const _Float16* w, int n0, int k0, int lane) {
    int n  = n0 + (lane & 15);
    int kb = k0 + ((lane & 16) ? 16 : 0);
    return *(const v16h*)(w + n * 64 + kb);   // 16 contiguous K values (32B)
}

__device__ __forceinline__ float fsig(float x) {
    return __fdividef(1.0f, 1.0f + __expf(-x));
}
__device__ __forceinline__ float ftanh(float x) {
    return 2.0f * fsig(2.0f * x) - 1.0f;      // v_exp + v_rcp fast path
}

// ---------------------------------------------------------------------------
// Kernel 0: cast all layers' W_ih (f32 -> f16), vectorized x4
// ---------------------------------------------------------------------------
__global__ __launch_bounds__(256) void k_castw(const float* __restrict__ w,
                                               _Float16* __restrict__ w16) {
    int i = (blockIdx.x * 256 + threadIdx.x) * 4;
    v4f v = *(const v4f*)(w + i);
    v4h h;
#pragma unroll
    for (int j = 0; j < 4; ++j) h[j] = (_Float16)v[j];
    *(v4h*)(w16 + i) = h;
}

// ---------------------------------------------------------------------------
// Kernel 1: embedding gather + cast to f16, time-major x[T][B][E], float4 loads
// ---------------------------------------------------------------------------
__global__ __launch_bounds__(256) void k_embed(const int* __restrict__ tokens,
                                               const float* __restrict__ emb,
                                               _Float16* __restrict__ x) {
    int idx = blockIdx.x * 256 + threadIdx.x;   // over T*B*16
    int q   = idx & 15;                         // quad within E
    int tb  = idx >> 4;                         // t*B + b
    int b   = tb & (BB - 1);
    int t   = tb >> 8;
    int tok = tokens[b * TT + t];               // tokens layout [B,T]
    v4f v = *(const v4f*)(emb + tok * EE + q * 4);
    v4h h;
#pragma unroll
    for (int j = 0; j < 4; ++j) h[j] = (_Float16)v[j];
    *(v4h*)(x + tb * EE + q * 4) = h;
}

// ---------------------------------------------------------------------------
// Kernel 2: input projection  xg = x[M,64] @ Wi^T + bi + bh  (M = T*B)
// xg is stored in WMMA C-tile layout: [tile = mtile*16+nt][lane][8 floats]
// so the scan can reload accumulators with contiguous 32B v8f loads.
// One block = 8 waves handles one 16-row M tile across all 16 N tiles.
// ---------------------------------------------------------------------------
__global__ __launch_bounds__(256) void k_inproj(const _Float16* __restrict__ x,
                                                const _Float16* __restrict__ Wi16,
                                                const float* __restrict__ bi,
                                                const float* __restrict__ bh,
                                                float* __restrict__ xg) {
    int m0   = blockIdx.x * 16;
    int wave = threadIdx.x >> 5;
    int lane = threadIdx.x & 31;

    // A tiles depend only on kc: load once, reuse across n-tiles
    v16h a0 = loadA16(x, m0, 0,  lane);
    v16h a1 = loadA16(x, m0, 32, lane);

#pragma unroll
    for (int rep = 0; rep < 2; ++rep) {
        int nt = wave + rep * 8;       // 16 N tiles over 8 waves
        int n0 = nt * 16;
        int nc = n0 + (lane & 15);
        float bias = bi[nc] + bh[nc];
        v8f acc;
#pragma unroll
        for (int j = 0; j < 8; ++j) acc[j] = bias;

        v16h b0 = loadBT16(Wi16, n0, 0,  lane);
        acc = __builtin_amdgcn_wmma_f32_16x16x32_f16(false, a0, false, b0,
                                                     (short)0, acc, false, false);
        v16h b1 = loadBT16(Wi16, n0, 32, lane);
        acc = __builtin_amdgcn_wmma_f32_16x16x32_f16(false, a1, false, b1,
                                                     (short)0, acc, false, false);

        // tile-layout store: one 32B store per lane, 1KB contiguous per wave
        ((v8f*)xg)[((size_t)blockIdx.x * 16 + nt) * 32 + lane] = acc;
    }
}

// ---------------------------------------------------------------------------
// Kernel 3: persistent recurrent scan for one layer (single workgroup).
// 1024 threads = 32 waves; each wave owns 2 combos (mTile in [0,16), hTile in
// [0,4)); per step: 4 gate tiles x K=64 -> 8 wmma per combo, gates in f32,
// c held in VGPRs, h f16 double-buffered in LDS -> ONE barrier per step.
// xnext copy of step t-1's h overlaps with step t's GEMM (both read sH[rb]).
// ---------------------------------------------------------------------------
__global__ __launch_bounds__(1024) void k_scan(const float* __restrict__ xg,
                                               const float* __restrict__ h0,
                                               const float* __restrict__ c0,
                                               const float* __restrict__ Wh,
                                               _Float16* __restrict__ xnext,
                                               float* __restrict__ hT_out,
                                               float* __restrict__ cT_out) {
    __shared__ _Float16 sWh[G4H * HH];      // 32 KB, recurrent weights f16
    __shared__ _Float16 sH[2][BB * HH];     // 2 x 32 KB, h ping-pong

    int tid  = threadIdx.x;
    int lane = tid & 31;
    int wave = tid >> 5;

    for (int i = tid; i < G4H * HH; i += 1024) {
        sWh[i]   = (_Float16)Wh[i];
        sH[0][i] = (_Float16)h0[i];         // both are 256*64 elements
    }

    // combo geometry + c registers
    int mT[2], hT[2];
    v8f cReg[2];
#pragma unroll
    for (int ci = 0; ci < 2; ++ci) {
        int cIdx = wave * 2 + ci;           // 0..63
        mT[ci] = cIdx >> 2;                 // batch tile 0..15
        hT[ci] = cIdx & 3;                  // h tile 0..3
        int b0 = mT[ci] * 16 + ((lane & 16) ? 8 : 0);
        int hc = hT[ci] * 16 + (lane & 15);
#pragma unroll
        for (int j = 0; j < 8; ++j) cReg[ci][j] = c0[(b0 + j) * HH + hc];
    }
    __syncthreads();

    for (int t = 0; t < TT; ++t) {
        int rb = t & 1;                     // read buffer (holds h_t)
        const _Float16* hRd = sH[rb];
        _Float16*       hWr = sH[rb ^ 1];
        v8f acc[2][4];

        // ---- phase 1a: stream previous step's h out to xnext (read||read) --
        if (t > 0) {
            const v8h* src = (const v8h*)hRd;                        // 2048 v8h
            v8h*       dst = (v8h*)(xnext + (size_t)(t - 1) * BB * HH);
            int base = tid * 2;
            dst[base]     = src[base];
            dst[base + 1] = src[base + 1];
        }

        // ---- phase 1b: recurrent GEMM (reads hRd) ----
#pragma unroll
        for (int ci = 0; ci < 2; ++ci) {
            int m0  = mT[ci] * 16;
            int hc0 = hT[ci] * 16;
            // accumulator init: contiguous v8f from tile-layout xg
#pragma unroll
            for (int g = 0; g < 4; ++g) {
                size_t tile = (size_t)(t * 16 + mT[ci]) * 16 + (g * 4 + hT[ci]);
                acc[ci][g] = ((const v8f*)xg)[tile * 32 + lane];
            }
#pragma unroll
            for (int kc = 0; kc < 64; kc += 32) {
                v16h a = loadA16(hRd, m0, kc, lane);
#pragma unroll
                for (int g = 0; g < 4; ++g) {
                    v16h b = loadBT16(sWh, g * HH + hc0, kc, lane);
                    acc[ci][g] = __builtin_amdgcn_wmma_f32_16x16x32_f16(
                        false, a, false, b, (short)0, acc[ci][g], false, false);
                }
            }
        }

        // prefetch next step's 256KB xg slab (2 cachelines per thread;
        // global_prefetch_b8 -- no VGPR dest, no counter)
        if (t + 1 < TT) {
            const char* nb = (const char*)xg + (size_t)(t + 1) * BB * G4H * 4;
            __builtin_prefetch(nb + tid * 256, 0, 3);
            __builtin_prefetch(nb + tid * 256 + 128, 0, 3);
        }

        // ---- phase 2: gates, update c (regs) and h (write buffer) ----
#pragma unroll
        for (int ci = 0; ci < 2; ++ci) {
            int b0 = mT[ci] * 16 + ((lane & 16) ? 8 : 0);
            int hc = hT[ci] * 16 + (lane & 15);
#pragma unroll
            for (int j = 0; j < 8; ++j) {
                float ig = fsig(acc[ci][0][j]);
                float fg = fsig(acc[ci][1][j]);
                float gg = ftanh(acc[ci][2][j]);
                float og = fsig(acc[ci][3][j]);
                float cn = fg * cReg[ci][j] + ig * gg;
                float hn = og * ftanh(cn);
                cReg[ci][j] = cn;
                int b = b0 + j;
                hWr[b * HH + hc] = (_Float16)hn;
                if (t == TT - 1) {
                    hT_out[b * HH + hc] = hn;
                    cT_out[b * HH + hc] = cn;
                }
            }
        }
        __syncthreads();   // single barrier: hWr visible for next phase 1
    }

    // final h (written at t=511 into sH[0]) -> xnext[511]
    {
        const v8h* src = (const v8h*)sH[0];
        v8h*       dst = (v8h*)(xnext + (size_t)(TT - 1) * BB * HH);
        int base = tid * 2;
        dst[base]     = src[base];
        dst[base + 1] = src[base + 1];
    }
}

// ---------------------------------------------------------------------------
// Kernel 4: out[b][o] = hT_last[b] . W_fc[o] + b_fc[o]
// ---------------------------------------------------------------------------
__global__ __launch_bounds__(256) void k_fc(const float* __restrict__ hT,
                                            const float* __restrict__ Wfc,
                                            const float* __restrict__ bfc,
                                            float* __restrict__ out) {
    int b = threadIdx.x;
#pragma unroll
    for (int o = 0; o < OO; ++o) {
        float s = bfc[o];
#pragma unroll 8
        for (int h = 0; h < HH; ++h) s += hT[b * HH + h] * Wfc[o * HH + h];
        out[b * OO + o] = s;
    }
}

// ---------------------------------------------------------------------------
extern "C" void kernel_launch(void* const* d_in, const int* in_sizes, int n_in,
                              void* d_out, int out_size, void* d_ws, size_t ws_size,
                              hipStream_t stream) {
    (void)in_sizes; (void)n_in; (void)out_size; (void)ws_size;

    const int*   tokens = (const int*)d_in[0];
    const float* h0     = (const float*)d_in[1];   // [L,B,H]
    const float* c0     = (const float*)d_in[2];
    const float* emb    = (const float*)d_in[3];   // [V,E]
    const float* Wih    = (const float*)d_in[4];   // [L,4H,E]
    const float* Whh    = (const float*)d_in[5];   // [L,4H,H]
    const float* bih    = (const float*)d_in[6];   // [L,4H]
    const float* bhh    = (const float*)d_in[7];
    const float* Wfc    = (const float*)d_in[8];   // [O,H]
    const float* bfc    = (const float*)d_in[9];

    float* out = (float*)d_out;                    // [B*O | L*B*H | L*B*H]
    float* hTs = out + BB * OO;
    float* cTs = hTs + LL * BB * HH;

    // Workspace: two f16 x buffers (16 MB each) + f16 W_ih (96 KB) + f32 xg
    char* ws = (char*)d_ws;
    _Float16* xa   = (_Float16*)ws;
    _Float16* xb   = xa + (size_t)TT * BB * EE;
    _Float16* wi16 = xb + (size_t)TT * BB * EE;
    float*    xg   = (float*)(wi16 + (size_t)LL * G4H * EE);

    const size_t layW = (size_t)G4H * EE;          // 256*64 per layer

    k_castw<<<(LL * G4H * EE) / (256 * 4), 256, 0, stream>>>(Wih, wi16);
    k_embed<<<(TT * BB * 16) / 256, 256, 0, stream>>>(tokens, emb, xa);

    _Float16* xin  = xa;
    _Float16* xout = xb;
    for (int l = 0; l < LL; ++l) {
        k_inproj<<<(TT * BB) / 16, 256, 0, stream>>>(
            xin, wi16 + layW * l, bih + (size_t)G4H * l, bhh + (size_t)G4H * l, xg);
        k_scan<<<1, 1024, 0, stream>>>(
            xg, h0 + (size_t)BB * HH * l, c0 + (size_t)BB * HH * l,
            Whh + layW * l, xout,
            hTs + (size_t)BB * HH * l, cTs + (size_t)BB * HH * l);
        _Float16* tmp = xin; xin = xout; xout = tmp;
    }

    k_fc<<<1, 256, 0, stream>>>(hTs + (size_t)BB * HH * (LL - 1), Wfc, bfc, out);
}
